// SSDLite_55894704390345
// MI455X (gfx1250) — matrix operand order repaired and checked by reading, not compiled, not run on hardware
//
#include <hip/hip_runtime.h>
#include <hip/hip_bf16.h>

// ---------------------------------------------------------------------------
// SSDLite detection head on gfx1250 (CDNA5, wave32, WMMA + async-LDS).
//
//   0) transpose_feat : [B,C,S,S] f32 -> [B,S,S,C] f16   (channel-last)
//   0) pack_weights   : loc+conf OIHW f32 -> Bpack[N_PAD][9*Cin] f16,
//                       K ordered as tap*Cin+ci; bias fused to [N_PAD].
//   1) head_gemm      : im2col GEMM.  Per workgroup (4 waves): B panel
//                       double-buffered in LDS via global_load_async_to_lds_b128
//                       (ASYNCcnt), each wave = 16 M-rows x 160 N-cols.
//                       B fragments are software-pipelined one tile deep so
//                       each v_wmma_f32_16x16x32_f16 overlaps the next tile's
//                       ds_load_b128 pair (wait dscnt<=2, not 0).
//   2) softmax_decode : softmax over 21 classes + SSD box decode.
//   3) nms_topk       : per-(image,class) top-200 + greedy NMS in LDS.
// ---------------------------------------------------------------------------

#define NUM_CLASSES 21
#define MBOX        6
#define TOP_K       200
#define CONF_T      0.01f
#define NMS_T       0.45f
#define BATCH       64
#define PRIORS      3234          // 6 * (400+100+25+9+4+1)
#define O_LOC       24            // MBOX*4
#define O_ALL       150           // MBOX*4 + MBOX*21
#define N_PAD       160           // padded to 16 -> 10 N-tiles
#define PANEL_H     (N_PAD * 32)  // halves per K-chunk panel (160 rows x 32 k)

typedef __attribute__((ext_vector_type(16))) _Float16 v16h;
typedef __attribute__((ext_vector_type(8)))  _Float16 v8h;
typedef __attribute__((ext_vector_type(8)))  float    v8f;

// ---------------------------------------------------------------------------
// Prep A: feature transpose + f32->f16 (channel-last, coalesced writes).
// ---------------------------------------------------------------------------
__global__ __launch_bounds__(256)
void transpose_feat(const float* __restrict__ feat,   // [B,C,S,S]
                    _Float16* __restrict__ featT,     // [B,S,S,C]
                    int Cin, int ss, int total)
{
    int o = blockIdx.x * blockDim.x + threadIdx.x;
    if (o >= total) return;
    const int ci   = o % Cin;
    const int rest = o / Cin;
    const int yx   = rest % ss;
    const int b    = rest / ss;
    featT[o] = (_Float16)feat[((size_t)b * Cin + ci) * ss + yx];
}

// ---------------------------------------------------------------------------
// Prep B: pack loc(24)+conf(126) weights into [N_PAD][9*Cin] f16 with
// k = tap*Cin + ci.  Padding rows are zero.  Bias fused to [N_PAD].
// ---------------------------------------------------------------------------
__global__ __launch_bounds__(256)
void pack_weights(const float* __restrict__ lw, const float* __restrict__ lb,
                  const float* __restrict__ cw, const float* __restrict__ cb,
                  _Float16* __restrict__ Bpack,       // [N_PAD, 9*Cin]
                  float* __restrict__ biasbuf,        // [N_PAD]
                  int Cin)
{
    const int K9 = 9 * Cin;
    int idx = blockIdx.x * blockDim.x + threadIdx.x;
    if (idx >= N_PAD * K9) return;
    const int n   = idx / K9;
    const int k   = idx - n * K9;
    const int tap = k / Cin;            // dy*3+dx
    const int ci  = k - tap * Cin;
    float v = 0.0f;
    if (n < O_LOC)      v = lw[((size_t)n * Cin + ci) * 9 + tap];
    else if (n < O_ALL) v = cw[((size_t)(n - O_LOC) * Cin + ci) * 9 + tap];
    Bpack[idx] = (_Float16)v;
    if (k == 0)
        biasbuf[n] = (n < O_LOC) ? lb[n] : (n < O_ALL ? cb[n - O_LOC] : 0.0f);
}

// ---------------------------------------------------------------------------
// Kernel 1: GEMM [M=B*S*S] x [K=9*Cin] x [N=160].
// Block = 4 waves, wave w -> M-tile (blockIdx.x*4+w), all 10 N-tiles.
// B panel per 32-wide K-chunk staged to LDS by async loads, double buffered.
// ---------------------------------------------------------------------------
__global__ __launch_bounds__(128)
void head_gemm(const _Float16* __restrict__ featT,   // [B,S,S,Cin] f16
               const _Float16* __restrict__ Bpack,   // [N_PAD, 9*Cin] f16
               const float* __restrict__ biasbuf,    // [N_PAD]
               float* __restrict__ locbuf,           // [B,PRIORS,4]
               float* __restrict__ confbuf,          // [B,PRIORS,21]
               int Cin, int S, int prior_off)
{
    __shared__ _Float16 sB[2][PANEL_H];              // 2 x 10 KB panels

    const int wave = threadIdx.x >> 5;
    const int lane = threadIdx.x & 31;
    const int hi   = lane >> 4;
    const int l15  = lane & 15;
    const int ss   = S * S;
    const int K9   = 9 * Cin;
    const int nchunks = K9 >> 5;                     // K-chunks of 32

    // A row this lane covers (M = lane&15 for both halves)
    const int m0  = (blockIdx.x * 4 + wave) * 16;
    const int mA  = m0 + l15;
    const int ab  = mA / ss;
    const int ayx = mA - ab * ss;
    const int aoy = ayx / S;
    const int aox = ayx - aoy * S;
    const _Float16* fb = featT + (size_t)ab * ss * Cin;

    const unsigned ldsbase = (unsigned)(uintptr_t)(&sB[0][0]);

    // ---- async stage of one K-chunk panel: 640 b128 blocks / 128 threads = 5
    auto stage = [&](int kc, int buf) {
        const size_t kofs = (size_t)kc * 32;
        #pragma unroll
        for (int i = 0; i < 5; ++i) {
            const int e  = threadIdx.x + 128 * i;    // b128 block id [0,640)
            const int n  = e >> 2;                   // panel row
            const int jb = e & 3;                    // 8-half block in row
            const unsigned long long ga =
                (unsigned long long)(const void*)(Bpack + (size_t)n * K9 + kofs + jb * 8);
            const unsigned dst = ldsbase + (unsigned)(buf * (PANEL_H * 2) + e * 16);
            asm volatile("global_load_async_to_lds_b128 %0, %1, off"
                         :: "v"(dst), "v"(ga) : "memory");
        }
    };

    // ---- one B fragment (16x32 slice of tile tt) from the LDS panel
    auto loadfrag = [&](int buf, int tt) -> v16h {
        const int off = (tt * 16 + l15) * 32 + hi * 16;
        const v8h b1 = *(const v8h*)&sB[buf][off];
        const v8h b2 = *(const v8h*)&sB[buf][off + 8];
        return __builtin_shufflevector(b1, b2,
                0, 1, 2, 3, 4, 5, 6, 7, 8, 9, 10, 11, 12, 13, 14, 15);
    };

    v8f acc[10];
    #pragma unroll
    for (int t = 0; t < 10; ++t) acc[t] = (v8f){};

    stage(0, 0);                                     // prime buffer 0

    int kc = 0;
    for (int tap = 0; tap < 9; ++tap) {
        const int dy = tap / 3, dx = tap - dy * 3;
        const int iy = aoy + dy - 1, ix = aox + dx - 1;
        const bool valid = ((unsigned)iy < (unsigned)S) & ((unsigned)ix < (unsigned)S);
        const _Float16* arow = fb + ((size_t)iy * S + ix) * Cin;

        for (int ci0 = 0; ci0 < Cin; ci0 += 32, ++kc) {
            const int buf = kc & 1;

            // ---- A fragment first: its global latency overlaps the barrier.
            //      halves {ci0+hi*16 ..+16} (contiguous)
            v16h av = {};
            if (valid) {
                const v8h r1 = *(const v8h*)(arow + ci0 + hi * 16);
                const v8h r2 = *(const v8h*)(arow + ci0 + hi * 16 + 8);
                av = __builtin_shufflevector(r1, r2,
                        0, 1, 2, 3, 4, 5, 6, 7, 8, 9, 10, 11, 12, 13, 14, 15);
            }

            if (kc + 1 < nchunks) {
                stage(kc + 1, buf ^ 1);              // prefetch next panel
                asm volatile("s_wait_asynccnt 0x5" ::: "memory");  // panel kc done
            } else {
                asm volatile("s_wait_asynccnt 0x0" ::: "memory");
            }
            __syncthreads();                         // publish panel kc

            // ---- 10 WMMAs, B fragments pipelined one tile ahead:
            //      ds_load(t+1) issued before wmma(t) -> wait dscnt<=2.
            v16h curf = loadfrag(buf, 0);
            #pragma unroll
            for (int t = 0; t < 10; ++t) {
                v16h nxtf;
                if (t < 9) nxtf = loadfrag(buf, t + 1);
                acc[t] = __builtin_amdgcn_wmma_f32_16x16x32_f16(
                             false, av, false, curf, (short)0, acc[t],
                             false, false);
                if (t < 9) curf = nxtf;
            }
            __syncthreads();         // all waves done with buf before re-stage
        }
    }

    // ---- scatter D tiles (N = lane&15; VGPR r -> M = r + hi*8) into SSD layout
    #pragma unroll
    for (int t = 0; t < 10; ++t) {
        const int n = t * 16 + l15;
        if (n >= O_ALL) continue;
        const float bias = biasbuf[n];
        #pragma unroll
        for (int r = 0; r < 8; ++r) {
            const int m   = m0 + r + hi * 8;
            const int bb  = m / ss;
            const int pyx = m - bb * ss;
            const float val = acc[t][r] + bias;
            if (n < O_LOC) {
                const int mbox = n >> 2, coord = n & 3;
                const int p = prior_off + pyx * MBOX + mbox;
                locbuf[((size_t)bb * PRIORS + p) * 4 + coord] = val;
            } else {
                const int j = n - O_LOC;
                const int mbox = j / NUM_CLASSES;
                const int cls  = j - mbox * NUM_CLASSES;
                const int p = prior_off + pyx * MBOX + mbox;
                confbuf[((size_t)bb * PRIORS + p) * NUM_CLASSES + cls] = val;
            }
        }
    }
}

// ---------------------------------------------------------------------------
// Kernel 2: softmax over classes + SSD decode. One thread per (b, prior).
// ---------------------------------------------------------------------------
__global__ __launch_bounds__(256)
void softmax_decode(const float* __restrict__ confbuf,   // [B,P,21]
                    const float* __restrict__ locbuf,    // [B,P,4]
                    const float* __restrict__ priors,    // [P,4] center-size
                    float* __restrict__ scores,          // [B,20,P]
                    float* __restrict__ boxes)           // [B,P,4] corner
{
    const int idx = blockIdx.x * blockDim.x + threadIdx.x;
    if (idx >= BATCH * PRIORS) return;
    const int b = idx / PRIORS;
    const int p = idx - b * PRIORS;

    const float* cf = confbuf + (size_t)idx * NUM_CLASSES;
    float mx = cf[0];
    #pragma unroll
    for (int c = 1; c < NUM_CLASSES; ++c) mx = fmaxf(mx, cf[c]);
    float e[NUM_CLASSES], sum = 0.0f;
    #pragma unroll
    for (int c = 0; c < NUM_CLASSES; ++c) { e[c] = __expf(cf[c] - mx); sum += e[c]; }
    const float inv = 1.0f / sum;
    #pragma unroll
    for (int c = 1; c < NUM_CLASSES; ++c)
        scores[((size_t)b * (NUM_CLASSES - 1) + (c - 1)) * PRIORS + p] = e[c] * inv;

    const float* lc = locbuf + (size_t)idx * 4;
    const float pcx = priors[p * 4 + 0], pcy = priors[p * 4 + 1];
    const float pw  = priors[p * 4 + 2], ph  = priors[p * 4 + 3];
    const float cx = pcx + lc[0] * 0.1f * pw;
    const float cy = pcy + lc[1] * 0.1f * ph;
    const float w  = pw * __expf(lc[2] * 0.2f);
    const float h  = ph * __expf(lc[3] * 0.2f);
    float* bx = boxes + (size_t)idx * 4;
    bx[0] = cx - 0.5f * w;
    bx[1] = cy - 0.5f * h;
    bx[2] = cx + 0.5f * w;
    bx[3] = cy + 0.5f * h;
}

// ---------------------------------------------------------------------------
// Kernel 3: per-(image,class) top-200 (iterative tie-broken argmax == sorted
// top_k) and greedy NMS in LDS.  One 256-thread block per (b, class).
// ---------------------------------------------------------------------------
__global__ __launch_bounds__(256)
void nms_topk(const float* __restrict__ scores,   // [B,20,P]
              const float* __restrict__ boxes,    // [B,P,4]
              float* __restrict__ out)            // [B,21,TOP_K,5]
{
    const int b = blockIdx.x / NUM_CLASSES;
    const int c = blockIdx.x % NUM_CLASSES;
    float* o = out + (size_t)blockIdx.x * TOP_K * 5;

    if (c == 0) {   // background slot: zeros (d_out is poisoned, must clear)
        for (int i = threadIdx.x; i < TOP_K * 5; i += blockDim.x) o[i] = 0.0f;
        return;
    }

    __shared__ float s_sc[PRIORS];
    __shared__ float sel_s[TOP_K];
    __shared__ float sel_b[TOP_K][4];
    __shared__ int   s_keep[TOP_K];
    __shared__ float rv[256];
    __shared__ int   ri[256];

    const float* sc  = scores + ((size_t)b * (NUM_CLASSES - 1) + (c - 1)) * PRIORS;
    const float* bxs = boxes + (size_t)b * PRIORS * 4;

    for (int i = threadIdx.x; i < PRIORS; i += blockDim.x) s_sc[i] = sc[i];
    __syncthreads();

    for (int t = 0; t < TOP_K; ++t) {
        float best = -3.402823466e+38f;
        int   bi   = 0x7fffffff;
        for (int i = threadIdx.x; i < PRIORS; i += blockDim.x) {
            const float v = s_sc[i];
            if (v > best || (v == best && i < bi)) { best = v; bi = i; }
        }
        rv[threadIdx.x] = best;
        ri[threadIdx.x] = bi;
        __syncthreads();
        for (int st = 128; st > 0; st >>= 1) {
            if (threadIdx.x < st) {
                const float v2 = rv[threadIdx.x + st];
                const int   i2 = ri[threadIdx.x + st];
                if (v2 > rv[threadIdx.x] ||
                    (v2 == rv[threadIdx.x] && i2 < ri[threadIdx.x])) {
                    rv[threadIdx.x] = v2;
                    ri[threadIdx.x] = i2;
                }
            }
            __syncthreads();
        }
        if (threadIdx.x == 0) {
            const int w = ri[0];
            sel_s[t]    = rv[0];
            sel_b[t][0] = bxs[w * 4 + 0];
            sel_b[t][1] = bxs[w * 4 + 1];
            sel_b[t][2] = bxs[w * 4 + 2];
            sel_b[t][3] = bxs[w * 4 + 3];
            s_sc[w]     = -3.402823466e+38f;
        }
        __syncthreads();
    }

    for (int i = threadIdx.x; i < TOP_K; i += blockDim.x)
        s_keep[i] = (sel_s[i] > CONF_T) ? 1 : 0;
    __syncthreads();

    for (int i = 0; i < TOP_K - 1; ++i) {
        if (s_keep[i]) {   // shared value -> block-uniform branch
            const float x1 = sel_b[i][0], y1 = sel_b[i][1];
            const float x2 = sel_b[i][2], y2 = sel_b[i][3];
            const float ai = (x2 - x1) * (y2 - y1);
            for (int j = i + 1 + threadIdx.x; j < TOP_K; j += blockDim.x) {
                const float u1 = fmaxf(x1, sel_b[j][0]);
                const float v1 = fmaxf(y1, sel_b[j][1]);
                const float u2 = fminf(x2, sel_b[j][2]);
                const float v2 = fminf(y2, sel_b[j][3]);
                const float iw = fmaxf(u2 - u1, 0.0f);
                const float ih = fmaxf(v2 - v1, 0.0f);
                const float inter = iw * ih;
                const float aj = (sel_b[j][2] - sel_b[j][0]) * (sel_b[j][3] - sel_b[j][1]);
                const float iou = inter / (ai + aj - inter + 1e-9f);
                if (iou > NMS_T) s_keep[j] = 0;
            }
        }
        __syncthreads();
    }

    for (int t = threadIdx.x; t < TOP_K; t += blockDim.x) {
        const float kf = s_keep[t] ? 1.0f : 0.0f;
        o[t * 5 + 0] = sel_s[t] * kf;
        o[t * 5 + 1] = sel_b[t][0] * kf;
        o[t * 5 + 2] = sel_b[t][1] * kf;
        o[t * 5 + 3] = sel_b[t][2] * kf;
        o[t * 5 + 4] = sel_b[t][3] * kf;
    }
}

// ---------------------------------------------------------------------------
extern "C" void kernel_launch(void* const* d_in, const int* in_sizes, int n_in,
                              void* d_out, int out_size, void* d_ws, size_t ws_size,
                              hipStream_t stream)
{
    static const int FM_C[6] = {576, 1280, 512, 256, 256, 64};
    static const int FM_S[6] = {20, 10, 5, 3, 2, 1};

    // ---- workspace carve -------------------------------------------------
    float* locbuf  = (float*)d_ws;
    float* confbuf = locbuf  + (size_t)BATCH * PRIORS * 4;
    float* scores  = confbuf + (size_t)BATCH * PRIORS * NUM_CLASSES;
    float* boxes   = scores  + (size_t)BATCH * (NUM_CLASSES - 1) * PRIORS;
    float* biasbuf = boxes   + (size_t)BATCH * PRIORS * 4;          // 6*N_PAD floats
    _Float16* hbase = (_Float16*)(biasbuf + 6 * N_PAD);

    _Float16* Bpack[6];
    _Float16* featT[6];
    _Float16* hp = hbase;
    for (int i = 0; i < 6; ++i) { Bpack[i] = hp; hp += (size_t)N_PAD * 9 * FM_C[i]; }
    for (int i = 0; i < 6; ++i) { featT[i] = hp; hp += (size_t)BATCH * FM_S[i] * FM_S[i] * FM_C[i]; }

    const float* priors = (const float*)d_in[30];

    // ---- prep: weight pack + feature transpose ---------------------------
    for (int i = 0; i < 6; ++i) {
        const float* lw = (const float*)d_in[5 * i + 1];
        const float* lb = (const float*)d_in[5 * i + 2];
        const float* cw = (const float*)d_in[5 * i + 3];
        const float* cb = (const float*)d_in[5 * i + 4];
        const int K9 = 9 * FM_C[i];
        const int nw = N_PAD * K9;
        pack_weights<<<(nw + 255) / 256, 256, 0, stream>>>(
            lw, lb, cw, cb, Bpack[i], biasbuf + i * N_PAD, FM_C[i]);

        const float* feat = (const float*)d_in[5 * i + 0];
        const int ss = FM_S[i] * FM_S[i];
        const int tot = BATCH * ss * FM_C[i];
        transpose_feat<<<(tot + 255) / 256, 256, 0, stream>>>(
            feat, featT[i], FM_C[i], ss, tot);
    }

    // ---- GEMM heads: 4 M-tiles per block (tilesM always divisible by 4) --
    int prior_off = 0;
    for (int i = 0; i < 6; ++i) {
        const int S = FM_S[i];
        const int tilesM = BATCH * S * S / 16;
        const int blocks = tilesM / 4;
        head_gemm<<<blocks, 128, 0, stream>>>(
            featT[i], Bpack[i], biasbuf + i * N_PAD,
            locbuf, confbuf, FM_C[i], S, prior_off);
        prior_off += MBOX * S * S;
    }

    // ---- tail ------------------------------------------------------------
    const int npix = BATCH * PRIORS;
    softmax_decode<<<(npix + 255) / 256, 256, 0, stream>>>(
        confbuf, locbuf, priors, scores, boxes);

    nms_topk<<<BATCH * NUM_CLASSES, 256, 0, stream>>>(scores, boxes, (float*)d_out);
}
